// BSpline_7395933684390
// MI455X (gfx1250) — matrix operand order repaired and checked by reading, not compiled, not run on hardware
//
#include <hip/hip_runtime.h>

// ---------------------------------------------------------------------------
// KAN layer as one folded WMMA GEMM on gfx1250 (wave32, v_wmma_f32_16x16x32_f16)
//   out[b,o] = sum_i ( sum_{n<11} basis_n(tanh(x[b,i])) * coeffs[i,o,n]
//                      + x[b,i] * base_weight[i,o] )
// K dimension = IN_F * 12  (11 spline rows + 1 residual row per feature).
// B is pre-packed to f16 [OUT_F][KDIM]; B tiles are staged into LDS by the
// Tensor Data Mover (tensor_load_to_lds, double-buffered, TENSORcnt-tracked)
// when the builtin is available, else by synchronous vector loads.
// This toolchain exposes the 6-argument tensor_load_to_lds builtin
// (g0 u32x4, g1 i32x8, g2 i32x4, g3 i32x4, extra i32x8, cpol i32).
// ---------------------------------------------------------------------------

#define IN_F   1024
#define OUT_F  1024
#define NCTRL  11
#define KPI    12                 // 11 ctrl rows + 1 residual row per feature
#define KDIM   (IN_F * KPI)       // 12288
#define BM     128
#define BN     128
#define BK     96                 // 8 features per K-block (8*12), 3 wmma k-steps
#define IPB    (BK / KPI)         // 8
#define NKB    (KDIM / BK)        // 128

#if defined(__has_builtin)
#  if __has_builtin(__builtin_amdgcn_tensor_load_to_lds) && \
      __has_builtin(__builtin_amdgcn_s_wait_tensorcnt)
#    define KAN_TDM 1
#  endif
#endif
#ifndef KAN_TDM
#  define KAN_TDM 0
#endif

#if KAN_TDM
#  define NBUF 2
#else
#  define NBUF 1
#endif

typedef _Float16     v8h   __attribute__((ext_vector_type(8)));
typedef _Float16     v16h  __attribute__((ext_vector_type(16)));
typedef float        v8f   __attribute__((ext_vector_type(8)));
typedef unsigned int u32x4 __attribute__((ext_vector_type(4)));
typedef int          i32x8 __attribute__((ext_vector_type(8)));
typedef int          i32x4 __attribute__((ext_vector_type(4)));

static __device__ __forceinline__ unsigned int packh2(float a, float b) {
  unsigned short ua = __builtin_bit_cast(unsigned short, (_Float16)a);
  unsigned short ub = __builtin_bit_cast(unsigned short, (_Float16)b);
  return (unsigned int)ua | ((unsigned int)ub << 16);
}

#if KAN_TDM
// byte offset of a shared-memory object within the LDS aperture
static __device__ __forceinline__ unsigned lds_offset_of(const void* p) {
  return (unsigned)(unsigned long long)
         (__attribute__((address_space(3))) const void*)p;
}

// Issue one TDM load of a 128-row x 96-half f16 tile (row stride KDIM halves)
// from global 'gsrc' into LDS byte offset 'ldsOff'.  D# per ISA ch.8:
//   group0: count=1 | lds_addr | global_addr[56:0] | type=2
//   group1: data_size=2B, tensor dims huge (no OOB clamp), tile=96x128,
//           tensor_dim0_stride=KDIM
//   groups 2/3 (and the trailing descriptor group): zero (2-D tile)
static __device__ __forceinline__ void tdm_load_tile(const _Float16* gsrc,
                                                     unsigned ldsOff) {
  unsigned long long ga = (unsigned long long)gsrc;
  u32x4 g0 = { 1u,                                   // count=1 (valid D#)
               ldsOff,                                // lds_addr
               (unsigned)(ga & 0xFFFFFFFFull),        // global_addr[31:0]
               (unsigned)((ga >> 32) & 0x1FFFFFFull)  // global_addr[56:32]
                 | 0x80000000u };                     // type=2 ("image")
  i32x8 g1 = { (int)0x00010000u,                      // data_size=1 (2 bytes)
               (int)0xFFFF0000u,                      // tensor_dim0 lo16<<16
               (int)0xFFFF7FFFu,                      // td0 hi | td1 lo<<16
               (int)((96u << 16) | 0x7FFFu),          // td1 hi | tile_dim0=96
               128,                                   // tile_dim1=128, tile_dim2=0
               (int)(unsigned)KDIM,                   // tensor_dim0_stride lo32
               0,                                     // stride0 hi | stride1 lo
               0 };                                   // stride1 hi
  i32x4 z4 = { 0, 0, 0, 0 };
  i32x8 z8 = { 0, 0, 0, 0, 0, 0, 0, 0 };
  __builtin_amdgcn_tensor_load_to_lds(g0, g1, z4, z4, z8, 0);
}
#endif

// ---------------------------------------------------------------------------
// Pack B = [coeffs interleaved with base_weight] -> f16, layout [OUT_F][KDIM]
// k = i*12 + n :  n<11 -> coeffs[i,o,n] ;  n==11 -> base_weight[i,o]
// ---------------------------------------------------------------------------
__global__ __launch_bounds__(256) void kan_pack_b(const float* __restrict__ coeffs,
                                                  const float* __restrict__ bw,
                                                  _Float16* __restrict__ Bp) {
  size_t idx = (size_t)blockIdx.x * 256 + threadIdx.x;   // exactly OUT_F*KDIM threads
  int o = (int)(idx / KDIM);
  int k = (int)(idx % KDIM);
  int i = k / KPI;
  int n = k % KPI;
  float v = (n < NCTRL) ? coeffs[((size_t)i * OUT_F + o) * NCTRL + n]
                        : bw[(size_t)i * OUT_F + o];
  Bp[idx] = (_Float16)v;
}

// ---------------------------------------------------------------------------
// Tiled WMMA GEMM: M=batch, N=OUT_F, K=KDIM. A (bases) computed on the fly.
// 256 threads = 8 waves in a 4(M) x 2(N) grid; wave tile 32x64 = 2x4 frags.
// ---------------------------------------------------------------------------
__global__ __launch_bounds__(256) void kan_wmma_gemm(const float* __restrict__ x,
                                                     const _Float16* __restrict__ Bp,
                                                     float* __restrict__ out) {
  __shared__ _Float16 sA[BM * BK];         // [row][k] row-major, 24 KB
  __shared__ _Float16 sB[NBUF][BN * BK];   // [col][k] col(N)-major, 24 KB each

  const int tid   = threadIdx.x;
  const int lane  = tid & 31;
  const int wave  = tid >> 5;        // 0..7
  const int waveM = wave & 3;        // 4 along M
  const int waveN = wave >> 2;       // 2 along N
  const int l16   = lane & 15;
  const int lhi   = lane >> 4;       // 0 or 1

  const int bm0 = blockIdx.y * BM;
  const int bn0 = blockIdx.x * BN;
  const _Float16* BpCol = Bp + (size_t)bn0 * KDIM;   // column-block base

  v8f acc[2][4] = {};

#if KAN_TDM
  int buf = 0;
  if (wave == 0) tdm_load_tile(BpCol, lds_offset_of(&sB[0][0]));
#endif

  for (int kb = 0; kb < NKB; ++kb) {
    const int i0 = kb * IPB;   // first input feature of this K-block
    const int k0 = kb * BK;

    __syncthreads();           // previous tiles fully consumed

    // ---- stage A: tanh + closed-form cubic B-spline bases -> LDS (f16) ----
    #pragma unroll
    for (int s = 0; s < 4; ++s) {
      int task = tid + (s << 8);          // 0..1023 = 128 rows x 8 features
      int r    = task >> 3;
      int il   = task & 7;
      float xv = x[(size_t)(bm0 + r) * IN_F + (i0 + il)];
      float t  = tanhf(xv);
      float sg = (t + 1.75f) * 4.0f;      // (t - grid[0]) / h
      float jf = floorf(sg);
      jf = fminf(fmaxf(jf, 3.0f), 10.0f); // knot interval (handles t == +/-1)
      float u  = sg - jf;
      int   nb = (int)jf - 3;             // first nonzero basis index, 0..7
      float um = 1.0f - u;
      float u2 = u * u, u3 = u2 * u;
      float w0 = um * um * um * (1.0f / 6.0f);
      float w1 = (3.0f * u3 - 6.0f * u2 + 4.0f) * (1.0f / 6.0f);
      float w2 = (-3.0f * u3 + 3.0f * u2 + 3.0f * u + 1.0f) * (1.0f / 6.0f);
      float w3 = u3 * (1.0f / 6.0f);
      float bb[KPI];
      #pragma unroll
      for (int n = 0; n < NCTRL; ++n) {
        float w = 0.0f;
        w = (n == nb    ) ? w0 : w;
        w = (n == nb + 1) ? w1 : w;
        w = (n == nb + 2) ? w2 : w;
        w = (n == nb + 3) ? w3 : w;
        bb[n] = w;
      }
      bb[NCTRL] = xv;                     // residual row
      unsigned int* d32 = (unsigned int*)(sA + r * BK + il * KPI); // 4B aligned
      #pragma unroll
      for (int p = 0; p < KPI / 2; ++p)
        d32[p] = packh2(bb[2 * p], bb[2 * p + 1]);
    }

    // hint next x block toward the caches (global_prefetch_b8)
    if (kb + 1 < NKB)
      __builtin_prefetch(x + (size_t)(bm0 + (tid >> 1)) * IN_F + (i0 + IPB), 0, 1);

#if KAN_TDM
    // ---- stage B: TDM double-buffer; issue kb+1, ensure kb has landed ----
    if (wave == 0) {
      if (kb + 1 < NKB) {
        tdm_load_tile(BpCol + (k0 + BK), lds_offset_of(&sB[buf ^ 1][0]));
        __builtin_amdgcn_s_wait_tensorcnt(1);   // tile kb complete (in-order)
      } else {
        __builtin_amdgcn_s_wait_tensorcnt(0);   // last tile complete
      }
    }
    __syncthreads();
    const _Float16* sBk = sB[buf];
#else
    // ---- stage B (fallback): pre-packed f16 [N][K] -> LDS, 16B loads ----
    #pragma unroll
    for (int c = 0; c < 6; ++c) {
      int chunk = tid + (c << 8);         // 0..1535 = 128 cols x 12 chunks
      int n  = chunk / 12;
      int kc = (chunk % 12) << 3;         // 8 halves per chunk
      *(v8h*)(sB[0] + n * BK + kc) =
          *(const v8h*)(BpCol + (size_t)n * KDIM + (k0 + kc));
    }
    __syncthreads();
    const _Float16* sBk = sB[0];
#endif

    // ---- compute: 3 k-steps of 32 ----
    #pragma unroll
    for (int ks = 0; ks < BK / 32; ++ks) {
      v16h af[2];
      v16h bf[4];
      // A frag (16x32 f16): lane holds row l16; K chunk pair by lane hi bit
      #pragma unroll
      for (int mi = 0; mi < 2; ++mi) {
        const _Float16* ap =
            sA + (waveM * 32 + mi * 16 + l16) * BK + ks * 32 + lhi * 8;
        v8h lo = *(const v8h*)ap;          // K = kb..kb+7
        v8h hi = *(const v8h*)(ap + 16);   // K = kb+16..kb+23
        af[mi] = __builtin_shufflevector(lo, hi,
                 0,1,2,3,4,5,6,7,8,9,10,11,12,13,14,15);
      }
      // B frag (32x16 f16): lane holds col l16; K chunk pair by lane hi bit
      #pragma unroll
      for (int ni = 0; ni < 4; ++ni) {
        const _Float16* bp =
            sBk + (waveN * 64 + ni * 16 + l16) * BK + ks * 32 + lhi * 8;
        v8h lo = *(const v8h*)bp;
        v8h hi = *(const v8h*)(bp + 16);
        bf[ni] = __builtin_shufflevector(lo, hi,
                 0,1,2,3,4,5,6,7,8,9,10,11,12,13,14,15);
      }
      #pragma unroll
      for (int mi = 0; mi < 2; ++mi)
        #pragma unroll
        for (int ni = 0; ni < 4; ++ni)
          acc[mi][ni] = __builtin_amdgcn_wmma_f32_16x16x32_f16(
              false, af[mi], false, bf[ni], (short)0, acc[mi][ni],
              false, false);
    }

#if KAN_TDM
    buf ^= 1;
#endif
  }

  // ---- epilogue: C layout = lane l16 -> col, VGPR v -> row v + 8*lhi ----
  #pragma unroll
  for (int mi = 0; mi < 2; ++mi) {
    #pragma unroll
    for (int ni = 0; ni < 4; ++ni) {
      int row0 = bm0 + waveM * 32 + mi * 16 + lhi * 8;
      int col  = bn0 + waveN * 64 + ni * 16 + l16;
      #pragma unroll
      for (int v = 0; v < 8; ++v)
        out[(size_t)(row0 + v) * OUT_F + col] = acc[mi][ni][v];
    }
  }
}

// ---------------------------------------------------------------------------
extern "C" void kernel_launch(void* const* d_in, const int* in_sizes, int n_in,
                              void* d_out, int out_size, void* d_ws, size_t ws_size,
                              hipStream_t stream) {
  const float* x      = (const float*)d_in[0];
  const float* coeffs = (const float*)d_in[1];
  const float* bw     = (const float*)d_in[2];
  // d_in[3] is the knot grid; it is the fixed uniform grid (h=0.25,
  // grid[0]=-1.75) baked into the kernels analytically.

  _Float16* Bp = (_Float16*)d_ws;          // needs OUT_F*KDIM*2 = 24 MiB
  (void)ws_size; (void)n_in; (void)out_size;

  int packBlocks = (OUT_F * KDIM) / 256;   // 49152
  kan_pack_b<<<packBlocks, 256, 0, stream>>>(coeffs, bw, Bp);

  int batch = in_sizes[0] / IN_F;          // 4096
  dim3 grid(OUT_F / BN, batch / BM);       // (8, 32)
  kan_wmma_gemm<<<grid, 256, 0, stream>>>(x, Bp, (float*)d_out);
}
